// MLA_27565100105886
// MI455X (gfx1250) — matrix operand order repaired and checked by reading, not compiled, not run on hardware
//
#include <hip/hip_runtime.h>

// ---------------------------------------------------------------------------
// MLA forward for MI455X (gfx1250, wave32, WMMA).
// All matmuls (projections + attention) run on v_wmma_f32_16x16x32_bf16.
// All global->LDS tile movement uses the gfx1250 async engine (ASYNCcnt),
// double-buffered in the GEMM and overlapped with QK^T in attention.
// Weights are transposed at convert time and V is produced transposed, so
// every WMMA fragment load is a contiguous ds_load_b128 (no u16 gathers).
// ~78 GFLOP total, ~100MB of traffic (fits in 192MB L2) -> matrix-op bound.
// ---------------------------------------------------------------------------

#define SEQ     2048
#define DMODEL  2048
#define NH      16
#define DH      128
#define DCQ     768
#define DCKV    512
#define DHR     64
#define DQK     192   // DH + DHR

typedef __bf16 bf16;
typedef __attribute__((ext_vector_type(16))) __bf16 v16bf;
typedef __attribute__((ext_vector_type(8)))  float  v8f;

static __device__ inline v8f wmma_bf16(v16bf a, v16bf b, v8f c) {
  // (neg_a, A, neg_b, B, c_mod, C, reuse_a, reuse_b)
  return __builtin_amdgcn_wmma_f32_16x16x32_bf16(false, a, false, b, (short)0, c,
                                                 false, false);
}

// Per-lane 16B global->LDS async copy (GLOBAL_LOAD_ASYNC_TO_LDS_B128,
// tracked by ASYNCcnt). LDS operand = low 32 bits of the flat address
static __device__ inline void async_copy_b128(void* lds_dst, const void* gsrc) {
  unsigned int lds_off = (unsigned int)(unsigned long long)lds_dst;
  asm volatile("global_load_async_to_lds_b128 %0, %1, off"
               :: "v"(lds_off), "v"(gsrc)
               : "memory");
}
#define WAIT_ASYNCCNT(n) asm volatile("s_wait_asynccnt %0" :: "n"(n) : "memory")

// ---------------------------------------------------------------------------
// fp32 -> bf16 convert with transpose: w (KxN, row) -> wt (NxK, row).
// Tiled through LDS so both global sides stay coalesced. block (32,8).
// ---------------------------------------------------------------------------
__global__ __launch_bounds__(256) void cvt_tr_f32_bf16(const float* __restrict__ w,
                                                       bf16* __restrict__ wt,
                                                       int K, int N) {
  __shared__ float tile[32][33];
  const int tx = threadIdx.x, ty = threadIdx.y;
  const int k0 = blockIdx.y * 32, n0 = blockIdx.x * 32;
#pragma unroll
  for (int j = 0; j < 4; ++j) {
    int r = ty + j * 8;
    tile[r][tx] = w[(size_t)(k0 + r) * N + n0 + tx];
  }
  __syncthreads();
#pragma unroll
  for (int j = 0; j < 4; ++j) {
    int r = ty + j * 8;
    wt[(size_t)(n0 + r) * K + k0 + tx] = (bf16)tile[tx][r];
  }
}

// ---------------------------------------------------------------------------
// RMSNorm: one row (2048) per block, 256 threads
// ---------------------------------------------------------------------------
__global__ __launch_bounds__(256) void rmsnorm_bf16(const float* __restrict__ x,
                                                    const float* __restrict__ w,
                                                    bf16* __restrict__ h) {
  const int row = blockIdx.x;
  const int tid = threadIdx.x;
  const int lane = tid & 31, wave = tid >> 5;
  const float* xr = x + (size_t)row * DMODEL;
  float ss = 0.f;
  for (int i = tid; i < DMODEL; i += 256) { float v = xr[i]; ss += v * v; }
#pragma unroll
  for (int off = 16; off > 0; off >>= 1) ss += __shfl_xor(ss, off, 32);
  __shared__ float red[8];
  if (lane == 0) red[wave] = ss;
  __syncthreads();
  if (tid == 0) {
    float t = 0.f;
#pragma unroll
    for (int i = 0; i < 8; ++i) t += red[i];
    red[0] = rsqrtf(t / (float)DMODEL + 1.1920929e-7f);
  }
  __syncthreads();
  const float rstd = red[0];
  bf16* hr = h + (size_t)row * DMODEL;
  for (int i = tid; i < DMODEL; i += 256) hr[i] = (bf16)(xr[i] * rstd * w[i]);
}

// ---------------------------------------------------------------------------
// Tiled bf16 WMMA GEMM: C(MxN) = A(MxK,row) * Bt(NxK,row)^T
// Block tile 128x64, 8 waves (4 in M x 2 in N), each wave 32x32 (4 accums).
// Double-buffered LDS tiles filled by async global->LDS DMA (3 async ops
// per thread per tile: 2 for A, 1 for Bt); s_wait_asynccnt 3 lets the next
// tile's DMA stay in flight while this tile is consumed by WMMA. Both A and
// B fragments are contiguous 32B per lane -> pure ds_load_b128 feeds.
// out_f32: fp32 output (+ optional residual). out_tr: write C^T (bf16).
// ---------------------------------------------------------------------------
__global__ __launch_bounds__(256) void gemm_bf16_wmma(
    const bf16* __restrict__ A, const bf16* __restrict__ Bt,
    bf16* __restrict__ Cb, float* __restrict__ Cf, const float* __restrict__ resid,
    int M, int N, int K, int out_f32, int out_tr)
{
  __shared__ __align__(16) bf16 sA[2][128][40];   // 128 rows x 32 k (+8 pad)
  __shared__ __align__(16) bf16 sB[2][64][40];    // 64 cols x 32 k (+8 pad)

  const int tid  = threadIdx.x;
  const int lane = tid & 31, wave = tid >> 5;
  const int n15  = lane & 15, hi = lane >> 4;
  const int wm   = wave & 3,  wn = wave >> 2;
  const int bm   = blockIdx.x * 128, bn = blockIdx.y * 64;

  v8f acc[2][2];
#pragma unroll
  for (int i = 0; i < 2; ++i)
#pragma unroll
    for (int j = 0; j < 2; ++j)
#pragma unroll
      for (int e = 0; e < 8; ++e) acc[i][j][e] = 0.f;

  const int ar = tid >> 1, ac = (tid & 1) * 16;   // A: 16 bf16 per thread (2 ops)
  const int br = tid >> 2, bc = (tid & 3) * 8;    // Bt: 8 bf16 per thread (1 op)

  const int ntiles = K >> 5;

  auto issue_tile = [&](int kt, int buf) {
    const int k0 = kt << 5;
    async_copy_b128(&sA[buf][ar][ac],     A  + (size_t)(bm + ar) * K + k0 + ac);
    async_copy_b128(&sA[buf][ar][ac + 8], A  + (size_t)(bm + ar) * K + k0 + ac + 8);
    async_copy_b128(&sB[buf][br][bc],     Bt + (size_t)(bn + br) * K + k0 + bc);
  };

  issue_tile(0, 0);

  for (int kt = 0; kt < ntiles; ++kt) {
    const int cur = kt & 1;
    if (kt + 1 < ntiles) {
      issue_tile(kt + 1, cur ^ 1);
      WAIT_ASYNCCNT(3);                 // tile kt complete; kt+1 in flight
    } else {
      WAIT_ASYNCCNT(0);
    }
    if (kt + 2 < ntiles)                // warm L2 two tiles ahead
      __builtin_prefetch(A + (size_t)(bm + ar) * K + ((kt + 2) << 5) + ac, 0, 1);
    __syncthreads();                    // all waves' DMA for tile kt visible

    v16bf afrag[2], bfrag[2];
#pragma unroll
    for (int mi = 0; mi < 2; ++mi) {
      const bf16* rp = &sA[cur][wm * 32 + mi * 16 + n15][0];
#pragma unroll
      for (int e = 0; e < 16; ++e) afrag[mi][e] = rp[16 * hi + e];
    }
#pragma unroll
    for (int ni = 0; ni < 2; ++ni) {
      const bf16* cp = &sB[cur][wn * 32 + ni * 16 + n15][16 * hi];
#pragma unroll
      for (int e = 0; e < 16; ++e) bfrag[ni][e] = cp[e];
    }
#pragma unroll
    for (int mi = 0; mi < 2; ++mi)
#pragma unroll
      for (int ni = 0; ni < 2; ++ni)
        acc[mi][ni] = wmma_bf16(afrag[mi], bfrag[ni], acc[mi][ni]);
    __syncthreads();                    // done reading buf before it is refilled
  }

  if (out_tr) {
    // C^T: each lane stores 8 contiguous bf16 per accumulator
#pragma unroll
    for (int mi = 0; mi < 2; ++mi)
#pragma unroll
      for (int ni = 0; ni < 2; ++ni) {
        int colg = bn + wn * 32 + ni * 16 + n15;
        int rowg = bm + wm * 32 + mi * 16 + 8 * hi;
        bf16* p = Cb + (size_t)colg * M + rowg;
#pragma unroll
        for (int r = 0; r < 8; ++r) p[r] = (bf16)acc[mi][ni][r];
      }
  } else {
#pragma unroll
    for (int mi = 0; mi < 2; ++mi)
#pragma unroll
      for (int ni = 0; ni < 2; ++ni)
#pragma unroll
        for (int r = 0; r < 8; ++r) {
          int row = bm + wm * 32 + mi * 16 + r + 8 * hi;
          int col = bn + wn * 32 + ni * 16 + n15;
          size_t idx = (size_t)row * N + col;
          float v = acc[mi][ni][r];
          if (out_f32) Cf[idx] = v + (resid ? resid[idx] : 0.f);
          else         Cb[idx] = (bf16)v;
        }
  }
}

// ---------------------------------------------------------------------------
// RoPE + pack into per-head [H][S][192] layouts. 64 threads, grid (S, NH).
// ---------------------------------------------------------------------------
__global__ __launch_bounds__(64) void pack_q_rope(const bf16* __restrict__ qc,
                                                  const bf16* __restrict__ qr,
                                                  bf16* __restrict__ qbuf) {
  const int s = blockIdx.x, h = blockIdx.y, t = threadIdx.x;
  bf16* dst = qbuf + ((size_t)h * SEQ + s) * DQK;
  const bf16* qcp = qc + (size_t)s * (NH * DH) + h * DH;
  dst[t] = qcp[t];
  dst[64 + t] = qcp[64 + t];
  if (t < 32) {
    const bf16* qrp = qr + (size_t)s * (NH * DHR) + h * DHR;
    float x0 = (float)qrp[2 * t], x1 = (float)qrp[2 * t + 1];
    float ang = (float)s * __powf(10000.f, -(float)t / 32.f);
    float sn, cs; __sincosf(ang, &sn, &cs);
    dst[DH + 2 * t]     = (bf16)(x0 * cs - x1 * sn);
    dst[DH + 2 * t + 1] = (bf16)(x0 * sn + x1 * cs);
  }
}

__global__ __launch_bounds__(64) void pack_k_rope(const bf16* __restrict__ kc,
                                                  const bf16* __restrict__ kr,
                                                  bf16* __restrict__ kbuf) {
  const int s = blockIdx.x, h = blockIdx.y, t = threadIdx.x;
  bf16* dst = kbuf + ((size_t)h * SEQ + s) * DQK;
  const bf16* kcp = kc + (size_t)s * (NH * DH) + h * DH;
  dst[t] = kcp[t];
  dst[64 + t] = kcp[64 + t];
  if (t < 32) {
    const bf16* krp = kr + (size_t)s * DHR;   // shared across heads
    float x0 = (float)krp[2 * t], x1 = (float)krp[2 * t + 1];
    float ang = (float)s * __powf(10000.f, -(float)t / 32.f);
    float sn, cs; __sincosf(ang, &sn, &cs);
    dst[DH + 2 * t]     = (bf16)(x0 * cs - x1 * sn);
    dst[DH + 2 * t + 1] = (bf16)(x0 * sn + x1 * cs);
  }
}

// ---------------------------------------------------------------------------
// Flash attention, causal, head dim 192 (QK) / 128 (V). Block = (qtile 64,
// head), 4 waves, each wave owns 16 query rows. Key blocks of 32.
// QK^T: 12 wmma / key block; PV: 8 wmma / key block (per wave).
// Q staged once; K and V tiles async-DMA'd per key block. V (from V^T) is
// issued after K so s_wait_asynccnt 4 releases QK^T while V is in flight.
// ---------------------------------------------------------------------------
__global__ __launch_bounds__(128) void mla_flash_attn(
    const bf16* __restrict__ qbuf,   // [H][S][192]
    const bf16* __restrict__ kbuf,   // [H][S][192]
    const bf16* __restrict__ vT,     // [H*128][S]  (V transposed)
    bf16* __restrict__ obuf)         // [S][H*128]
{
  __shared__ __align__(16) bf16 sQ[64][200];     // 64 x 192 (+8 pad)
  __shared__ __align__(16) bf16 sK[32][200];     // 32 keys x 192 (+8 pad)
  __shared__ __align__(16) bf16 sVt[128][40];    // [dv][key] (+8 pad)
  __shared__ __align__(16) bf16 sP[4][16][40];   // per-wave P (16 x 32)

  const int tid  = threadIdx.x;
  const int lane = tid & 31, wave = tid >> 5;
  const int n15  = lane & 15, hi = lane >> 4;
  const int head = blockIdx.y, qb = blockIdx.x;

  const bf16* Q   = qbuf + ((size_t)head * SEQ + (size_t)qb * 64) * DQK;
  const bf16* Kh  = kbuf + (size_t)head * SEQ * DQK;
  const bf16* vTh = vT + (size_t)head * DH * SEQ;

  // stage Q (64 x 192) via async DMA: 12 b128 chunks per thread
  for (int ch = tid; ch < 64 * (DQK / 8); ch += 128) {
    int r = ch / (DQK / 8), c8 = (ch % (DQK / 8)) * 8;
    async_copy_b128(&sQ[r][c8], Q + (size_t)r * DQK + c8);
  }
  WAIT_ASYNCCNT(0);
  __syncthreads();

  float m[8], l[8];
  v8f o[8];
#pragma unroll
  for (int r = 0; r < 8; ++r) { m[r] = -1e30f; l[r] = 0.f; }
#pragma unroll
  for (int t = 0; t < 8; ++t)
#pragma unroll
    for (int e = 0; e < 8; ++e) o[t][e] = 0.f;

  const int   qrow0 = qb * 64 + wave * 16;
  const float scale = 0.07216878364870322f;   // 1/sqrt(192)
  const int   nkb   = 2 * qb + 2;             // causal: keys <= tile end

  for (int kb = 0; kb < nkb; ++kb) {
    const int key0 = kb * 32;
    // K tile (32 x 192): 6 async ops per thread (issued first)
#pragma unroll
    for (int i = 0; i < 6; ++i) {
      int ch = tid + i * 128;
      int r = ch / 24, c8 = (ch % 24) * 8;
      async_copy_b128(&sK[r][c8], Kh + (size_t)(key0 + r) * DQK + c8);
    }
    // V tile (128 dv x 32 keys) from V^T: 4 async ops per thread
#pragma unroll
    for (int i = 0; i < 4; ++i) {
      int ch = tid + i * 128;
      int dv = ch >> 2, c = (ch & 3) * 8;
      async_copy_b128(&sVt[dv][c], vTh + (size_t)dv * SEQ + key0 + c);
    }
    WAIT_ASYNCCNT(4);    // K complete; V still in flight
    __syncthreads();

    // S = Q K^T (16 q rows x 32 keys)
    v8f s0, s1;
#pragma unroll
    for (int e = 0; e < 8; ++e) { s0[e] = 0.f; s1[e] = 0.f; }
#pragma unroll
    for (int d0 = 0; d0 < DQK; d0 += 32) {
      v16bf a;
      const bf16* rp = &sQ[wave * 16 + n15][d0];
#pragma unroll
      for (int e = 0; e < 16; ++e) a[e] = rp[16 * hi + e];
      v16bf b0, b1;
      const bf16* kr0 = &sK[n15][d0 + 16 * hi];
      const bf16* kr1 = &sK[16 + n15][d0 + 16 * hi];
#pragma unroll
      for (int e = 0; e < 16; ++e) { b0[e] = kr0[e]; b1[e] = kr1[e]; }
      s0 = wmma_bf16(a, b0, s0);
      s1 = wmma_bf16(a, b1, s1);
    }
    // online softmax (reductions stay within 16-lane C-layout groups)
#pragma unroll
    for (int r = 0; r < 8; ++r) {
      int qrow = qrow0 + r + 8 * hi;
      float v0 = s0[r] * scale, v1 = s1[r] * scale;
      if (key0 + n15 > qrow)      v0 = -1e30f;
      if (key0 + 16 + n15 > qrow) v1 = -1e30f;
      float mx = fmaxf(v0, v1);
#pragma unroll
      for (int off = 1; off < 16; off <<= 1) mx = fmaxf(mx, __shfl_xor(mx, off, 32));
      float mn = fmaxf(m[r], mx);
      float alpha = __expf(m[r] - mn);
      float p0 = __expf(v0 - mn), p1 = __expf(v1 - mn);
      float rs = p0 + p1;
#pragma unroll
      for (int off = 1; off < 16; off <<= 1) rs += __shfl_xor(rs, off, 32);
      m[r] = mn;
      l[r] = l[r] * alpha + rs;
#pragma unroll
      for (int t = 0; t < 8; ++t) o[t][r] *= alpha;
      sP[wave][r + 8 * hi][n15]      = (bf16)p0;
      sP[wave][r + 8 * hi][16 + n15] = (bf16)p1;
    }
    WAIT_ASYNCCNT(0);    // V tile landed (DMA overlapped QK^T + softmax)
    __syncthreads();     // sVt + sP visible
    // O += P @ V
    v16bf pa;
    {
      const bf16* rp = &sP[wave][n15][0];
#pragma unroll
      for (int e = 0; e < 16; ++e) pa[e] = rp[16 * hi + e];
    }
#pragma unroll
    for (int t = 0; t < 8; ++t) {
      v16bf b;
      const bf16* vp = &sVt[16 * t + n15][16 * hi];
#pragma unroll
      for (int e = 0; e < 16; ++e) b[e] = vp[e];
      o[t] = wmma_bf16(pa, b, o[t]);
    }
    __syncthreads();     // protect sK/sVt/sP before next iteration's DMA
  }

#pragma unroll
  for (int t = 0; t < 8; ++t)
#pragma unroll
    for (int r = 0; r < 8; ++r) {
      int row = qrow0 + r + 8 * hi;
      int col = head * DH + 16 * t + n15;
      obuf[(size_t)row * (NH * DH) + col] = (bf16)(o[t][r] / l[r]);
    }
}

// ---------------------------------------------------------------------------
// Host-side orchestration
// ---------------------------------------------------------------------------
extern "C" void kernel_launch(void* const* d_in, const int* in_sizes, int n_in,
                              void* d_out, int out_size, void* d_ws, size_t ws_size,
                              hipStream_t stream) {
  (void)in_sizes; (void)n_in; (void)out_size; (void)ws_size;
  const float* x      = (const float*)d_in[0];
  // d_in[1] = mask (bool) — causal mask computed analytically, not read
  const float* w_norm = (const float*)d_in[2];
  const float* w_cq   = (const float*)d_in[3];
  const float* w_q    = (const float*)d_in[4];
  const float* w_qr   = (const float*)d_in[5];
  const float* w_ckv  = (const float*)d_in[6];
  const float* w_k    = (const float*)d_in[7];
  const float* w_kr   = (const float*)d_in[8];
  const float* w_v    = (const float*)d_in[9];
  const float* w_o    = (const float*)d_in[10];
  float* out = (float*)d_out;

  char* ws = (char*)d_ws;
  size_t off = 0;
  auto alloc = [&](size_t elems) -> bf16* {
    bf16* p = (bf16*)(ws + off);
    off += (elems * sizeof(bf16) + 255) & ~(size_t)255;
    return p;
  };

  // transposed bf16 weights: wT is (N x K) row-major
  bf16* tw_cq  = alloc((size_t)DCQ * DMODEL);
  bf16* tw_q   = alloc((size_t)NH * DH * DCQ);
  bf16* tw_qr  = alloc((size_t)NH * DHR * DCQ);
  bf16* tw_ckv = alloc((size_t)DCKV * DMODEL);
  bf16* tw_k   = alloc((size_t)NH * DH * DCKV);
  bf16* tw_kr  = alloc((size_t)DHR * DMODEL);
  bf16* tw_v   = alloc((size_t)NH * DH * DCKV);
  bf16* tw_o   = alloc((size_t)DMODEL * NH * DH);
  bf16* h      = alloc((size_t)SEQ * DMODEL);
  bf16* c_q    = alloc((size_t)SEQ * DCQ);
  bf16* c_kv   = alloc((size_t)SEQ * DCKV);
  bf16* qcF    = alloc((size_t)SEQ * NH * DH);
  bf16* qrF    = alloc((size_t)SEQ * NH * DHR);
  bf16* kcF    = alloc((size_t)SEQ * NH * DH);
  bf16* krF    = alloc((size_t)SEQ * DHR);
  bf16* vT     = alloc((size_t)NH * DH * SEQ);
  bf16* qbuf   = alloc((size_t)NH * SEQ * DQK);
  bf16* kbuf   = alloc((size_t)NH * SEQ * DQK);
  bf16* obuf   = qcF;  // reuse: qcF dead after pack_q_rope

  // weight convert + transpose (w: KxN -> wT: NxK)
  struct { const float* s; bf16* d; int K, N; } cv[8] = {
    { w_cq,  tw_cq,  DMODEL, DCQ },      { w_q,  tw_q,  DCQ,  NH * DH },
    { w_qr,  tw_qr,  DCQ,  NH * DHR },   { w_ckv, tw_ckv, DMODEL, DCKV },
    { w_k,   tw_k,   DCKV, NH * DH },    { w_kr, tw_kr, DMODEL, DHR },
    { w_v,   tw_v,   DCKV, NH * DH },    { w_o,  tw_o,  NH * DH, DMODEL },
  };
  for (int i = 0; i < 8; ++i) {
    dim3 grid(cv[i].N / 32, cv[i].K / 32), blk(32, 8);
    cvt_tr_f32_bf16<<<grid, blk, 0, stream>>>(cv[i].s, cv[i].d, cv[i].K, cv[i].N);
  }

  rmsnorm_bf16<<<SEQ, 256, 0, stream>>>(x, w_norm, h);

  auto gemm = [&](const bf16* A, const bf16* Bt, bf16* Cb, float* Cf,
                  const float* resid, int M, int N, int K, int f32, int tr) {
    dim3 grid(M / 128, N / 64);
    gemm_bf16_wmma<<<grid, 256, 0, stream>>>(A, Bt, Cb, Cf, resid, M, N, K, f32, tr);
  };

  gemm(h,    tw_cq,  c_q,  nullptr, nullptr, SEQ, DCQ,      DMODEL, 0, 0);
  gemm(h,    tw_ckv, c_kv, nullptr, nullptr, SEQ, DCKV,     DMODEL, 0, 0);
  gemm(h,    tw_kr,  krF,  nullptr, nullptr, SEQ, DHR,      DMODEL, 0, 0);
  gemm(c_q,  tw_q,   qcF,  nullptr, nullptr, SEQ, NH * DH,  DCQ,    0, 0);
  gemm(c_q,  tw_qr,  qrF,  nullptr, nullptr, SEQ, NH * DHR, DCQ,    0, 0);
  gemm(c_kv, tw_k,   kcF,  nullptr, nullptr, SEQ, NH * DH,  DCKV,   0, 0);
  gemm(c_kv, tw_v,   vT,   nullptr, nullptr, SEQ, NH * DH,  DCKV,   0, 1);  // V^T

  {
    dim3 grid(SEQ, NH);
    pack_q_rope<<<grid, 64, 0, stream>>>(qcF, qrF, qbuf);
    pack_k_rope<<<grid, 64, 0, stream>>>(kcF, krF, kbuf);
  }

  {
    dim3 grid(SEQ / 64, NH);
    mla_flash_attn<<<grid, 128, 0, stream>>>(qbuf, kbuf, vT, obuf);
  }

  // out = obuf @ w_o + x  (fp32 output with residual)
  gemm(obuf, tw_o, nullptr, out, x, SEQ, DMODEL, NH * DH, 1, 0);
}
